// Attention_56521769615974
// MI455X (gfx1250) — compile-verified, matrix operand order checked by reading, not run on hardware
//
#include <hip/hip_runtime.h>
#include <hip/hip_bf16.h>

// ---------------------------------------------------------------------------
// MHA forward for B=8, N=1024, C=768, H=12, D=64 on gfx1250 (wave32, WMMA).
// Compute-bound (~65 GFLOP vs ~40MB traffic) -> everything through
// v_wmma_f32_16x16x32_bf16 with fp32 accumulation.
// Round 2: wave tile 32x64 (8 accums) + launch_bounds to kill scratch spills.
// ---------------------------------------------------------------------------

typedef unsigned short u16;
typedef unsigned int   u32;
typedef __attribute__((ext_vector_type(16))) __bf16 v16bf;
typedef __attribute__((ext_vector_type(8)))  float  v8f;

#define NHEAD   12
#define DHEAD   64
#define NSEQ    1024
#define NBATCH  8
#define CDIM    768
#define QKVDIM  2304
#define MROWS   (NBATCH * NSEQ)   // 8192
#define ATT_SCALE 0.125f          // 1/sqrt(64)

// round-to-nearest-even fp32 -> bf16
__device__ __forceinline__ u16 f2bf(float f) {
  u32 u = __builtin_bit_cast(u32, f);
  u32 r = u + 0x7FFFu + ((u >> 16) & 1u);
  return (u16)(r >> 16);
}

union FragB { v16bf v; u32 u[8]; };

// Load a 16x32 bf16 fragment whose "K" runs contiguously in memory.
// Works for A (row = M, storage row-major MxK) and for B when the B source is
// stored K-contiguous per output column (i.e. transposed weights / Kbf / Vt).
// Per-lane VGPR layout (ISA 7.12.2, 16-bit): lane = h*16 + r,
//   vgpr v in 0..3 -> K = h*8 + 2v{,+1};  vgpr v in 4..7 -> K = 16 + h*8 + 2(v-4){,+1}
__device__ __forceinline__ void load_frag_k(FragB& f, const u16* __restrict__ base,
                                            int row, int ld, int kBase, int lane) {
  const int r  = lane & 15;
  const int hh = lane >> 4;
  const u16* p = base + (size_t)(row + r) * (size_t)ld + kBase + hh * 8;
  f.u[0] = *(const u32*)(p + 0);
  f.u[1] = *(const u32*)(p + 2);
  f.u[2] = *(const u32*)(p + 4);
  f.u[3] = *(const u32*)(p + 6);
  f.u[4] = *(const u32*)(p + 16);
  f.u[5] = *(const u32*)(p + 18);
  f.u[6] = *(const u32*)(p + 20);
  f.u[7] = *(const u32*)(p + 22);
}

__device__ __forceinline__ v8f wmma_bf16(const FragB& a, const FragB& b, v8f c) {
  return __builtin_amdgcn_wmma_f32_16x16x32_bf16(false, a.v, false, b.v,
                                                 (short)0, c, false, false);
}

// ---------------------------------------------------------------------------
// fp32 -> bf16 elementwise convert
// ---------------------------------------------------------------------------
__global__ void cvt_f32_bf16_kernel(const float* __restrict__ in,
                                    u16* __restrict__ out, int n) {
  int i = blockIdx.x * 256 + threadIdx.x;
  if (i < n) out[i] = f2bf(in[i]);
}

// ---------------------------------------------------------------------------
// fp32 [rows][cols] -> bf16 [cols][rows] (transpose so K is contiguous per col)
// ---------------------------------------------------------------------------
__global__ void transpose_f32_bf16_kernel(const float* __restrict__ in,
                                          u16* __restrict__ out,
                                          int rows, int cols) {
  int i = blockIdx.x * 256 + threadIdx.x;
  if (i < rows * cols) {
    int r = i % rows;          // output writes coalesced along r
    int c = i / rows;
    out[(size_t)c * rows + r] = f2bf(in[(size_t)r * cols + c]);
  }
}

// ---------------------------------------------------------------------------
// bf16 GEMM: out[M,N] = A[M,K] * Bt[N,K]^T + bias[N]
// 256 threads = 8 waves, 4x2 wave grid, 32x64 tile per wave (2x4 accums,
// ~130 live VGPRs -> no spills). Block tile = 128x128.
// mode 0: fp32 output to outF
// mode 1: QKV epilogue -> scatter bf16 to Q [BH,N,D], K [BH,N,D], Vt [BH,D,N]
// ---------------------------------------------------------------------------
__global__ __launch_bounds__(256, 1)
void gemm_bf16_kernel(const u16* __restrict__ A,
                      const u16* __restrict__ Bt,
                      const float* __restrict__ bias,
                      int M, int N, int K, int mode,
                      float* __restrict__ outF,
                      u16* __restrict__ Qb,
                      u16* __restrict__ Kb,
                      u16* __restrict__ Vt) {
  const int tid  = threadIdx.x;
  const int lane = tid & 31;
  const int wid  = tid >> 5;
  const int wM   = wid & 3;        // 4 waves along M
  const int wN   = wid >> 2;       // 2 waves along N
  const int rowBase = blockIdx.x * 128 + wM * 32;
  const int colBase = blockIdx.y * 128 + wN * 64;

  v8f acc[2][4] = {};
  for (int k = 0; k < K; k += 32) {
    FragB a[2], b[4];
#pragma unroll
    for (int i = 0; i < 2; ++i) load_frag_k(a[i], A,  rowBase + i * 16, K, k, lane);
#pragma unroll
    for (int j = 0; j < 4; ++j) load_frag_k(b[j], Bt, colBase + j * 16, K, k, lane);
#pragma unroll
    for (int i = 0; i < 2; ++i)
#pragma unroll
      for (int j = 0; j < 4; ++j)
        acc[i][j] = wmma_bf16(a[i], b[j], acc[i][j]);
  }

  // C/D layout: lane = h*16 + n0; vgpr r -> row = r + 8*h, col = n0
  const int n0 = lane & 15;
  const int hh = lane >> 4;
#pragma unroll
  for (int i = 0; i < 2; ++i) {
#pragma unroll
    for (int j = 0; j < 4; ++j) {
#pragma unroll
      for (int r = 0; r < 8; ++r) {
        const int row = rowBase + i * 16 + r + 8 * hh;
        const int col = colBase + j * 16 + n0;
        const float val = acc[i][j][r] + bias[col];
        if (mode == 0) {
          outF[(size_t)row * N + col] = val;
        } else {
          // col in [0,2304): 0..767 Q, 768..1535 K, 1536..2303 V
          const int part = col / CDIM;
          const int c    = col % CDIM;
          const int h    = c / DHEAD;
          const int d    = c % DHEAD;
          const int bb   = row / NSEQ;
          const int n    = row % NSEQ;
          const int bh   = bb * NHEAD + h;
          const u16 bv   = f2bf(val);
          if (part == 0)      Qb[((size_t)bh * NSEQ + n) * DHEAD + d] = bv;
          else if (part == 1) Kb[((size_t)bh * NSEQ + n) * DHEAD + d] = bv;
          else                Vt[((size_t)bh * DHEAD + d) * NSEQ + n] = bv;
        }
      }
    }
  }
}

// ---------------------------------------------------------------------------
// Attention core. Grid (96, 32): (b*H+h, 32-query block). Block = 64 threads
// (2 waves, 16 query rows each).
// Phase 1: S[32][1024] = scale * Q Kt   (WMMA, fp32 in LDS, padded stride)
// Phase 2: softmax rows (unnormalized exp -> bf16 P, 1/rowsum kept aside)
// Phase 3: O[32][64] = P V via WMMA, scaled by 1/rowsum, stored bf16 [B,N,H*D]
// ---------------------------------------------------------------------------
#define SROW 1028   // fp32 stride (1024 + pad) -> avoids LDS bank aliasing
#define PROW 1032   // bf16 stride (1024 + pad)

__global__ __launch_bounds__(64, 1)
void attn_kernel(const u16* __restrict__ Q,
                 const u16* __restrict__ K,
                 const u16* __restrict__ Vt,
                 u16* __restrict__ Obf) {
  extern __shared__ char smem[];
  float* S    = (float*)smem;                                   // 32*SROW f32
  u16*   P    = (u16*)(smem + 32 * SROW * 4);                   // 32*PROW bf16
  float* red  = (float*)(smem + 32 * SROW * 4 + 32 * PROW * 2); // 64 partials
  float* invs = red + 64;                                       // 32 inv-sums

  const int bh   = blockIdx.x;                 // 0..95
  const int qblk = blockIdx.y;                 // 0..31
  const u16* Qb = Q  + (size_t)bh * NSEQ * DHEAD;
  const u16* Kb = K  + (size_t)bh * NSEQ * DHEAD;
  const u16* Vb = Vt + (size_t)bh * DHEAD * NSEQ;

  const int tid  = threadIdx.x;
  const int lane = tid & 31;
  const int wid  = tid >> 5;                   // 0..1
  const int qBase = qblk * 32 + wid * 16;      // global query row of this wave

  // ---- Phase 1: scores into LDS ----
  FragB qf0, qf1;
  load_frag_k(qf0, Qb, qBase, DHEAD, 0,  lane);
  load_frag_k(qf1, Qb, qBase, DHEAD, 32, lane);
  const int n0 = lane & 15;
  const int hh = lane >> 4;
  for (int kt = 0; kt < NSEQ / 16; ++kt) {
    FragB kf0, kf1;
    load_frag_k(kf0, Kb, kt * 16, DHEAD, 0,  lane);
    load_frag_k(kf1, Kb, kt * 16, DHEAD, 32, lane);
    v8f sacc = {};
    sacc = wmma_bf16(qf0, kf0, sacc);
    sacc = wmma_bf16(qf1, kf1, sacc);
    const int r0  = wid * 16 + hh * 8;
    const int col = kt * 16 + n0;
#pragma unroll
    for (int r = 0; r < 8; ++r)
      S[(size_t)(r0 + r) * SROW + col] = sacc[r] * ATT_SCALE;
  }
  __syncthreads();

  // ---- Phase 2: softmax (2 threads per row, 512 cols each) ----
  {
    const int row  = tid >> 1;
    const int half = tid & 1;
    const float* srow = S + (size_t)row * SROW + half * 512;
    float m = -1e30f;
    for (int i = 0; i < 512; ++i) m = fmaxf(m, srow[i]);
    red[tid] = m;
    __syncthreads();
    const float rowmax = fmaxf(red[2 * row], red[2 * row + 1]);
    __syncthreads();
    u16* prow = P + (size_t)row * PROW + half * 512;
    float s = 0.f;
    for (int i = 0; i < 512; ++i) {
      float e = __expf(srow[i] - rowmax);
      s += e;
      prow[i] = f2bf(e);          // unnormalized; 1/sum folded into PV epilogue
    }
    red[tid] = s;
    __syncthreads();
    if (half == 0) invs[row] = 1.0f / (red[2 * row] + red[2 * row + 1]);
  }
  __syncthreads();

  // ---- Phase 3: O = P * V ----
  v8f oacc[4] = {};
  const u16* Pw = P + (size_t)(wid * 16) * PROW;
  for (int kk = 0; kk < NSEQ; kk += 32) {
    FragB af;
    load_frag_k(af, Pw, 0, PROW, kk, lane);
#pragma unroll
    for (int dt = 0; dt < 4; ++dt) {
      FragB bfrag;
      load_frag_k(bfrag, Vb, dt * 16, NSEQ, kk, lane);
      oacc[dt] = wmma_bf16(af, bfrag, oacc[dt]);
    }
  }
  const int bb = bh / NHEAD;
  const int h  = bh % NHEAD;
#pragma unroll
  for (int dt = 0; dt < 4; ++dt) {
#pragma unroll
    for (int r = 0; r < 8; ++r) {
      const int qLocal  = wid * 16 + r + 8 * hh;    // 0..31
      const int qGlobal = qblk * 32 + qLocal;
      const int d       = dt * 16 + n0;
      const float val   = oacc[dt][r] * invs[qLocal];
      Obf[((size_t)(bb * NSEQ + qGlobal)) * CDIM + h * DHEAD + d] = f2bf(val);
    }
  }
}

// ---------------------------------------------------------------------------
// Host launcher
// ---------------------------------------------------------------------------
extern "C" void kernel_launch(void* const* d_in, const int* in_sizes, int n_in,
                              void* d_out, int out_size, void* d_ws, size_t ws_size,
                              hipStream_t stream) {
  const float* x     = (const float*)d_in[0];   // [8,1024,768]
  const float* W_qkv = (const float*)d_in[1];   // [768,2304]
  const float* b_qkv = (const float*)d_in[2];   // [2304]
  const float* W_out = (const float*)d_in[3];   // [768,768]
  const float* b_out = (const float*)d_in[4];   // [768]
  float* out = (float*)d_out;                   // [8,1024,768]

  u16* ws = (u16*)d_ws;
  const size_t N_X     = (size_t)MROWS * CDIM;        // 6291456
  const size_t N_WQKVT = (size_t)QKVDIM * CDIM;       // 1769472
  const size_t N_WOUTT = (size_t)CDIM * CDIM;         // 589824
  const size_t N_HEADS = (size_t)NBATCH * NHEAD * NSEQ * DHEAD; // 6291456

  u16* Xbf   = ws;
  u16* WqkvT = Xbf   + N_X;
  u16* WoutT = WqkvT + N_WQKVT;
  u16* Qb    = WoutT + N_WOUTT;
  u16* Kb    = Qb    + N_HEADS;
  u16* Vt    = Kb    + N_HEADS;
  u16* Obf   = Vt    + N_HEADS;

  // 1. precision conversion / weight transposition
  cvt_f32_bf16_kernel<<<(int)((N_X + 255) / 256), 256, 0, stream>>>(x, Xbf, (int)N_X);
  transpose_f32_bf16_kernel<<<(int)((N_WQKVT + 255) / 256), 256, 0, stream>>>(
      W_qkv, WqkvT, CDIM, QKVDIM);
  transpose_f32_bf16_kernel<<<(int)((N_WOUTT + 255) / 256), 256, 0, stream>>>(
      W_out, WoutT, CDIM, CDIM);

  // 2. QKV projection (M=8192, N=2304, K=768) with head-scatter epilogue
  gemm_bf16_kernel<<<dim3(MROWS / 128, QKVDIM / 128), 256, 0, stream>>>(
      Xbf, WqkvT, b_qkv, MROWS, QKVDIM, CDIM, /*mode=*/1,
      nullptr, Qb, Kb, Vt);

  // 3. attention core (96 head-batches x 32 query blocks)
  const size_t smemBytes = (size_t)32 * SROW * 4 + (size_t)32 * PROW * 2 + 96 * 4;
  (void)hipFuncSetAttribute((const void*)attn_kernel,
                            hipFuncAttributeMaxDynamicSharedMemorySize,
                            (int)smemBytes);
  attn_kernel<<<dim3(NBATCH * NHEAD, NSEQ / 32), 64, smemBytes, stream>>>(
      Qb, Kb, Vt, Obf);

  // 4. output projection (M=8192, N=768, K=768), fp32 out + bias
  gemm_bf16_kernel<<<dim3(MROWS / 128, CDIM / 128), 256, 0, stream>>>(
      Obf, WoutT, b_out, MROWS, CDIM, CDIM, /*mode=*/0,
      out, nullptr, nullptr, nullptr);
}
